// DotProductAttention_34651796144476
// MI455X (gfx1250) — matrix operand order repaired and checked by reading, not compiled, not run on hardware
//
#include <hip/hip_runtime.h>
#include <hip/hip_bf16.h>
#include <type_traits>

// ---------------------------------------------------------------------------
// CDNA5 (gfx1250) causal attention, bf16 WMMA with f32 accumulation.
// Workspace (~218 MB): Xbf [16384x1024], WqT/WkT/WvdT/WvupT (bf16 transposed),
// Qbf/Kbf/Tbf [16384x256] bf16, Vt [B][E][N] bf16, m/l row stats f32,
// P [B][N][N] bf16 (only the lower 128-aligned band region is written & read).
// ---------------------------------------------------------------------------

typedef __attribute__((ext_vector_type(16))) __bf16  v16bf;
typedef __attribute__((ext_vector_type(8)))  float   v8f;
typedef __attribute__((ext_vector_type(4)))  unsigned int u32x4;

#define NEG_FILL (-3.4028234663852886e+38f)

union V16U { v16bf v; u32x4 q[2]; };

__device__ __forceinline__ __bf16 f2bf(float f) {
  unsigned u = __builtin_bit_cast(unsigned, f);
  unsigned r = u + 0x7FFFu + ((u >> 16) & 1u);           // round-to-nearest-even
  unsigned short h = (unsigned short)(r >> 16);
  return __builtin_bit_cast(__bf16, h);
}

__device__ __forceinline__ v8f v8f_zero() {
  v8f z = {0.f, 0.f, 0.f, 0.f, 0.f, 0.f, 0.f, 0.f};
  return z;
}

// A-fragment, 16x32 bf16 (ISA 7.12.2 16-bit A layout):
// lane<16 : row=lane,    K {0..7} in elems 0..7, K {16..23} in elems 8..15
// lane>=16: row=lane-16, K {8..15},{24..31}
__device__ __forceinline__ v16bf load_a(const __bf16* base, int ld, int lane) {
  const int half = (lane >> 4) & 1, r = lane & 15;
  const __bf16* p = base + (size_t)r * ld + half * 8;
  V16U u;
  u.q[0] = *(const u32x4*)(p);
  u.q[1] = *(const u32x4*)(p + 16);
  return u.v;
}

// B-fragment, 32x16 bf16 from Bt (= B transposed, row-major [Ncols][Kdim]):
// lane&15 = column, lane<16: K {0..7},{16..23}; lane>=16: K {8..15},{24..31}.
__device__ __forceinline__ v16bf load_b(const __bf16* bt, int ld, int lane) {
  const int half = (lane >> 4) & 1, c = lane & 15;
  const __bf16* p = bt + (size_t)c * ld + half * 8;
  V16U u;
  u.q[0] = *(const u32x4*)(p);
  u.q[1] = *(const u32x4*)(p + 16);
  return u.v;
}

__device__ __forceinline__ v8f wmma_bf16(v16bf a, v16bf b, v8f c) {
  return __builtin_amdgcn_wmma_f32_16x16x32_bf16(false, a, false, b, (short)0, c,
                                                 false, false);
}

// ---------------------------------------------------------------------------
__global__ void cvt_f32_to_bf16_k(const float* __restrict__ src,
                                  __bf16* __restrict__ dst, int n) {
  int idx = blockIdx.x * blockDim.x + threadIdx.x;
  if (idx < n) dst[idx] = f2bf(src[idx]);
}

__global__ void transpose_f32_to_bf16_k(const float* __restrict__ src,
                                        __bf16* __restrict__ dst, int R, int C) {
  int idx = blockIdx.x * blockDim.x + threadIdx.x;
  if (idx < R * C) {
    int r = idx / C, c = idx % C;
    dst[(size_t)c * R + r] = f2bf(src[idx]);   // dst[C][R] = src^T
  }
}

// ---------------------------------------------------------------------------
// bf16 WMMA GEMM: C = A[M,K] * Bt[N,K]^T.  128 threads / 4 waves.
// Workgroup tile 128x64: each wave owns 32 rows (two A fragments sharing every
// B fragment -> 1.5 b128 loads per WMMA) x 64 cols (4 subtiles).
// TRANS: store C transposed ([col*M + row]) within each batch (for V^T).
// causal: limit K to blockIdx.x*128+128 (lower-band A, used for P@V).
template <typename TC, bool TRANS>
__global__ void __launch_bounds__(128)
gemm_wmma_k(const __bf16* __restrict__ A, const __bf16* __restrict__ Bt,
            TC* __restrict__ C, int M, int Ncols, int Kdim,
            long long sA, long long sB, long long sC, int causal) {
  const int z = blockIdx.z;
  const __bf16* Ab = A  + (long long)z * sA;
  const __bf16* Bb = Bt + (long long)z * sB;
  TC*           Cb = C  + (long long)z * sC;
  const int lane = threadIdx.x & 31;
  const int wave = threadIdx.x >> 5;
  const int half = (lane >> 4) & 1;
  const int l15  = lane & 15;
  const int row0 = blockIdx.x * 128 + wave * 32;
  const int col0 = blockIdx.y * 64;

  v8f acc[2][4];
#pragma unroll
  for (int r = 0; r < 2; ++r)
#pragma unroll
    for (int c = 0; c < 4; ++c) acc[r][c] = v8f_zero();

  int kend = Kdim;
  if (causal) { int lim = blockIdx.x * 128 + 128; kend = lim < Kdim ? lim : Kdim; }

  for (int k = 0; k < kend; k += 32) {
    if (k + 64 < kend)  // gfx1250 global_prefetch_b8 for the next A slab
      __builtin_prefetch(Ab + (size_t)row0 * Kdim + k + 64, 0, 1);
    v16bf a0 = load_a(Ab + (size_t)row0 * Kdim + k,        Kdim, lane);
    v16bf a1 = load_a(Ab + (size_t)(row0 + 16) * Kdim + k, Kdim, lane);
#pragma unroll
    for (int c = 0; c < 4; ++c) {
      v16bf b = load_b(Bb + (size_t)(col0 + 16 * c) * Kdim + k, Kdim, lane);
      acc[0][c] = wmma_bf16(a0, b, acc[0][c]);
      acc[1][c] = wmma_bf16(a1, b, acc[1][c]);
    }
  }

  // C/D fragment layout: VGPR i -> row = i + 8*half, col = lane&15.
#pragma unroll
  for (int rt = 0; rt < 2; ++rt) {
#pragma unroll
    for (int c = 0; c < 4; ++c) {
#pragma unroll
      for (int i = 0; i < 8; ++i) {
        const int r  = row0 + rt * 16 + i + half * 8;
        const int cc = col0 + c * 16 + l15;
        const float v = acc[rt][c][i];
        if constexpr (TRANS) {
          if constexpr (std::is_same<TC, float>::value) Cb[(size_t)cc * M + r] = v;
          else                                          Cb[(size_t)cc * M + r] = f2bf(v);
        } else {
          if constexpr (std::is_same<TC, float>::value) Cb[(size_t)r * Ncols + cc] = v;
          else                                          Cb[(size_t)r * Ncols + cc] = f2bf(v);
        }
      }
    }
  }
}

// ---------------------------------------------------------------------------
// Attention scores: S = (Q K^T)/16, causal mask, "exact zero -> FLT_MIN" quirk.
// Pass 1 (WRITE_P=false): online row max/sumexp -> mbuf/lbuf.
// Pass 2 (WRITE_P=true) : recompute S, write P = exp(S - m)/l as bf16.
// 4 waves x 16 query rows = 64-row band; key blocks of 32 are processed up to
// the 128-aligned band edge so the causal O-GEMM (128-row tiles) never reads
// uninitialized P.
template <bool WRITE_P>
__global__ void __launch_bounds__(128)
attn_scores_k(const __bf16* __restrict__ Q, const __bf16* __restrict__ K,
              float* __restrict__ mbuf, float* __restrict__ lbuf,
              __bf16* __restrict__ P) {
  const int N = 4096, D = 256;
  const int b    = blockIdx.y;
  const int lane = threadIdx.x & 31;
  const int wave = threadIdx.x >> 5;
  const int half = (lane >> 4) & 1;
  const int l15  = lane & 15;
  const int q0   = blockIdx.x * 64 + wave * 16;
  const __bf16* Qb = Q + (size_t)b * N * D;
  const __bf16* Kb = K + (size_t)b * N * D;

  v16bf qf[8];                               // 16 rows x full D=256, resident
#pragma unroll
  for (int t = 0; t < 8; ++t)
    qf[t] = load_a(Qb + (size_t)q0 * D + 32 * t, D, lane);

  float m[8], l[8];
#pragma unroll
  for (int i = 0; i < 8; ++i) {
    if constexpr (WRITE_P) {
      const int row = q0 + i + half * 8;
      m[i] = mbuf[b * N + row];
      l[i] = 1.0f / lbuf[b * N + row];       // reciprocal of sumexp
    } else { m[i] = -INFINITY; l[i] = 0.0f; }
  }

  // cover keys up to the 128-aligned band edge (zero-fill for the O-GEMM)
  const int nkb = ((blockIdx.x >> 1) + 1) * 4;
  for (int kbi = 0; kbi < nkb; ++kbi) {
    const int kb = kbi * 32;
    v8f s0 = v8f_zero(), s1 = v8f_zero();
#pragma unroll
    for (int t = 0; t < 8; ++t) {            // K row-major == Bt of K^T
      v16bf b0 = load_b(Kb + (size_t)kb * D + 32 * t, D, lane);
      v16bf b1 = load_b(Kb + (size_t)(kb + 16) * D + 32 * t, D, lane);
      s0 = wmma_bf16(qf[t], b0, s0);
      s1 = wmma_bf16(qf[t], b1, s1);
    }
#pragma unroll
    for (int i = 0; i < 8; ++i) {
      const int row = q0 + i + half * 8;
      const int c0  = kb + l15;
      const int c1  = kb + 16 + l15;
      float v0 = s0[i] * 0.0625f;            // 1/sqrt(256)
      float v1 = s1[i] * 0.0625f;
      v0 = (c0 <= row) ? v0 : NEG_FILL;      // causal: masked -> 0 -> NEG_FILL
      v1 = (c1 <= row) ? v1 : NEG_FILL;
      v0 = (v0 != 0.0f) ? v0 : NEG_FILL;     // reference quirk on exact zeros
      v1 = (v1 != 0.0f) ? v1 : NEG_FILL;
      if constexpr (WRITE_P) {
        const float e0 = __expf(v0 - m[i]) * l[i];
        const float e1 = __expf(v1 - m[i]) * l[i];
        const size_t base = (size_t)b * N * N + (size_t)row * N;
        P[base + c0] = f2bf(e0);
        P[base + c1] = f2bf(e1);
      } else {
        // row data lives in one 16-lane half; xor 8/4/2/1 stays inside it
        float bm = fmaxf(v0, v1);
#pragma unroll
        for (int o = 8; o >= 1; o >>= 1) bm = fmaxf(bm, __shfl_xor(bm, o, 32));
        const float nm = fmaxf(m[i], bm);
        float bs = __expf(v0 - nm) + __expf(v1 - nm);
#pragma unroll
        for (int o = 8; o >= 1; o >>= 1) bs += __shfl_xor(bs, o, 32);
        l[i] = l[i] * __expf(m[i] - nm) + bs;
        m[i] = nm;
      }
    }
  }

  if constexpr (!WRITE_P) {
    if (l15 == 0) {                          // lanes 0 / 16 publish rows 0-7 / 8-15
#pragma unroll
      for (int i = 0; i < 8; ++i) {
        const int row = q0 + i + half * 8;
        mbuf[b * N + row] = m[i];
        lbuf[b * N + row] = l[i];
      }
    }
  }
}

// ---------------------------------------------------------------------------
extern "C" void kernel_launch(void* const* d_in, const int* in_sizes, int n_in,
                              void* d_out, int out_size, void* d_ws, size_t ws_size,
                              hipStream_t stream) {
  (void)in_sizes; (void)n_in; (void)out_size; (void)ws_size;
  constexpr int B = 4, N = 4096, E = 1024, D = 256;
  constexpr int M = B * N;

  const float* X      = (const float*)d_in[0];
  const float* Wq     = (const float*)d_in[1];
  const float* Wk     = (const float*)d_in[2];
  const float* Wvdown = (const float*)d_in[3];
  const float* Wvup   = (const float*)d_in[4];
  float* out = (float*)d_out;

  char* ws = (char*)d_ws;
  size_t off = 0;
  auto alloc = [&](size_t bytes) -> void* {
    void* p = ws + off;
    off = (off + bytes + 255) & ~(size_t)255;
    return p;
  };
  __bf16* Xbf  = (__bf16*)alloc((size_t)M * E * 2);   // bf16 copy of inputs
  __bf16* WqT  = (__bf16*)alloc((size_t)D * E * 2);   // [D][E]
  __bf16* WkT  = (__bf16*)alloc((size_t)D * E * 2);
  __bf16* WvdT = (__bf16*)alloc((size_t)D * E * 2);
  __bf16* WvuT = (__bf16*)alloc((size_t)E * D * 2);   // [E][D]
  __bf16* Qbf  = (__bf16*)alloc((size_t)M * D * 2);
  __bf16* Kbf  = (__bf16*)alloc((size_t)M * D * 2);
  __bf16* Tbf  = (__bf16*)alloc((size_t)M * D * 2);   // X @ Wvdown
  __bf16* Vt   = (__bf16*)alloc((size_t)B * E * N * 2); // V^T per batch [E][N]
  float*  mbuf = (float*)alloc((size_t)M * 4);
  float*  lbuf = (float*)alloc((size_t)M * 4);
  __bf16* P    = (__bf16*)alloc((size_t)B * N * N * 2);

  dim3 tb(256);
  cvt_f32_to_bf16_k<<<(M * E + 255) / 256, tb, 0, stream>>>(X, Xbf, M * E);
  const int we = (E * D + 255) / 256;
  transpose_f32_to_bf16_k<<<we, tb, 0, stream>>>(Wq,     WqT,  E, D);
  transpose_f32_to_bf16_k<<<we, tb, 0, stream>>>(Wk,     WkT,  E, D);
  transpose_f32_to_bf16_k<<<we, tb, 0, stream>>>(Wvdown, WvdT, E, D);
  transpose_f32_to_bf16_k<<<we, tb, 0, stream>>>(Wvup,   WvuT, D, E);

  dim3 blk(128);
  // Projections: [16384,1024] @ [1024,256]
  gemm_wmma_k<__bf16, false><<<dim3(M / 128, D / 64, 1), blk, 0, stream>>>(
      Xbf, WqT,  Qbf, M, D, E, 0, 0, 0, 0);
  gemm_wmma_k<__bf16, false><<<dim3(M / 128, D / 64, 1), blk, 0, stream>>>(
      Xbf, WkT,  Kbf, M, D, E, 0, 0, 0, 0);
  gemm_wmma_k<__bf16, false><<<dim3(M / 128, D / 64, 1), blk, 0, stream>>>(
      Xbf, WvdT, Tbf, M, D, E, 0, 0, 0, 0);
  // V = T @ Wvup, stored transposed per batch as Vt[b][e][n]
  gemm_wmma_k<__bf16, true><<<dim3(N / 128, E / 64, B), blk, 0, stream>>>(
      Tbf, WvuT, Vt, N, E, D, (long long)N * D, 0, (long long)E * N, 0);
  // Softmax stats, then normalized P (bf16)
  attn_scores_k<false><<<dim3(N / 64, B), blk, 0, stream>>>(
      Qbf, Kbf, mbuf, lbuf, (__bf16*)nullptr);
  attn_scores_k<true><<<dim3(N / 64, B), blk, 0, stream>>>(
      Qbf, Kbf, mbuf, lbuf, P);
  // O = P @ V  (causal K-limit per 128-row band), f32 output
  gemm_wmma_k<float, false><<<dim3(N / 128, E / 64, B), blk, 0, stream>>>(
      P, Vt, out, N, E, N, (long long)N * N, (long long)E * N,
      (long long)N * E, 1);
}
